// Conv1x1Net_1340029796572
// MI455X (gfx1250) — matrix operand order repaired and checked by reading, not compiled, not run on hardware
//
#include <hip/hip_runtime.h>

// ---------------------------------------------------------------------------
// Problem constants (from reference)
// ---------------------------------------------------------------------------
#define D_LAYERS 8
#define BATCH    4
#define HW       16384          // 128*128
#define C        256            // CIN == COUT
#define WDIM     128
#define NEG_SLOPE 0.01f

// LDS B-tile: 128 rows (block N extent) x 256 K, f16, padded row pitch.
// pitch = 256 + 8 halves = 528 bytes -> bank stride 132 DW -> 4 banks/row.
#define BROWS   128
#define BPITCH  264             // halves per LDS row (256 data + 8 pad)

typedef _Float16 half8   __attribute__((ext_vector_type(8)));
typedef _Float16 half16  __attribute__((ext_vector_type(16)));
typedef float    float8  __attribute__((ext_vector_type(8)));
typedef unsigned int u32x4 __attribute__((ext_vector_type(4)));
typedef int          i32x4 __attribute__((ext_vector_type(4)));
typedef int          i32x8 __attribute__((ext_vector_type(8)));

// ---------------------------------------------------------------------------
// TDM: stage a 2D f16 tile (tile_d1 rows x tile_d0 elems, row stride d0_stride
// elems) from global into LDS at lds_byte, padding 4 DWORDs every 128 DWORDs
// (so 256-half rows land at BPITCH spacing). Issued by one wave; caller must
// s_wait_tensorcnt + barrier before consuming.
// ---------------------------------------------------------------------------
#if __has_builtin(__builtin_amdgcn_tensor_load_to_lds)
#define HAVE_TDM 1
__device__ __forceinline__ void tdm_load_b_tile(const void* gaddr, unsigned lds_byte) {
    unsigned long long ga = (unsigned long long)(uintptr_t)gaddr;
    u32x4 g0;
    g0.x = 1u;                                         // count=1 (valid D#)
    g0.y = lds_byte;                                   // lds_addr
    g0.z = (unsigned)(ga & 0xffffffffu);               // global_addr[31:0]
    g0.w = (unsigned)((ga >> 32) & 0x01ffffffu)        // global_addr[56:32]
         | 0x80000000u;                                // type=2 ("image")
    // group1: data_size=1 (2B), pad_enable, pad_interval=6 (128 DW=512B),
    // pad_amount=3 (4 DW = 8 halves)
    unsigned w0 = (1u << 16) | (1u << 20) | (6u << 22) | (3u << 25);
    i32x8 g1;
    g1[0] = (int)w0;
    g1[1] = (int)(256u << 16);                 // tensor_dim0 = 256 (lo16 @ [31:16])
    g1[2] = (int)(128u << 16);                 // tensor_dim0 hi=0 | tensor_dim1 lo = 128
    g1[3] = (int)(256u << 16);                 // tensor_dim1 hi=0 | tile_dim0 = 256
    g1[4] = (int)128u;                         // tile_dim1 = 128, tile_dim2 = 0
    g1[5] = (int)256u;                         // tensor_dim0_stride = 256 (lo32)
    g1[6] = 0;                                 // stride hi | dim1_stride lo
    g1[7] = 0;
    i32x4 gz = {0, 0, 0, 0};
#if __has_include(<hip/amd_detail/amd_gfx1250_TDM.h>)
    i32x8 gz8 = {0, 0, 0, 0, 0, 0, 0, 0};
    __builtin_amdgcn_tensor_load_to_lds(g0, g1, gz, gz, gz8, 0);   // clang-23 6-arg
#else
    __builtin_amdgcn_tensor_load_to_lds(g0, g1, gz, gz, 0);        // ROCm7.2 5-arg
#endif
}
#else
#define HAVE_TDM 0
#endif

// Stage B slice [row0..row0+127][0..255] of Wb (rows of length C) into Bs.
__device__ __forceinline__ void stage_B(const _Float16* __restrict__ Wbb,
                                        int row0, _Float16* Bs) {
#if HAVE_TDM
    if ((threadIdx.x >> 5) == 0) {
        tdm_load_b_tile(Wbb + (size_t)row0 * C,
                        (unsigned)(uintptr_t)(void*)Bs);  // low 32 bits = LDS offset
        __builtin_amdgcn_s_wait_tensorcnt(0);
    }
    __syncthreads();
#else
    // cooperative fallback: 256 threads copy 128x256 halves with padded pitch
    for (int t = threadIdx.x; t < BROWS * (C / 8); t += 256) {
        int r = t >> 5;            // row
        int c8 = (t & 31) * 8;     // col (in halves)
        *(half8*)&Bs[r * BPITCH + c8] = *(const half8*)&Wbb[((size_t)(row0 + r)) * C + c8];
    }
    __syncthreads();
#endif
}

// ---------------------------------------------------------------------------
// P0: f32 -> f16 convert
// ---------------------------------------------------------------------------
__global__ void cvt_f32_to_f16(const float* __restrict__ src,
                               _Float16* __restrict__ dst, int n) {
    int i = blockIdx.x * blockDim.x + threadIdx.x;
    if (i < n) dst[i] = (_Float16)src[i];
}

// ---------------------------------------------------------------------------
// P1: styles[l][b][i] = gain * dot(ws[l][b], affine_w[l][i]) + affine_b[l][i]
// ---------------------------------------------------------------------------
__global__ void styles_kernel(const float* __restrict__ wsv,   // [D][B][WDIM]
                              const float* __restrict__ aw,    // [D][C][WDIM]
                              const float* __restrict__ ab,    // [D][C]
                              float* __restrict__ styles) {    // [D][B][C]
    int idx = blockIdx.x * blockDim.x + threadIdx.x;
    if (idx >= D_LAYERS * BATCH * C) return;
    int i = idx % C;
    int b = (idx / C) % BATCH;
    int l = idx / (BATCH * C);
    const float* w = wsv + (l * BATCH + b) * WDIM;
    const float* a = aw + ((size_t)l * C + i) * WDIM;
    float acc = 0.f;
    for (int k = 0; k < WDIM; ++k) acc += w[k] * a[k];
    const float gain = 0.08838834764831845f;  // 1/sqrt(128)
    styles[idx] = acc * gain + ab[l * C + i];
}

// ---------------------------------------------------------------------------
// P2: demodulated per-(layer,batch) weights, f16, layout [l][b][o][i]
// ---------------------------------------------------------------------------
__global__ void modweights_kernel(const float* __restrict__ styles, // [D][B][C]
                                  const float* __restrict__ convw,  // [D][C][C]
                                  _Float16* __restrict__ wbH) {     // [D][B][C][C]
    int l = blockIdx.x;
    __shared__ float red[256];
    __shared__ float sN[BATCH * C];
    const float* st = styles + (size_t)l * BATCH * C;
    float ss = 0.f;
    for (int t = threadIdx.x; t < BATCH * C; t += 256) ss += st[t] * st[t];
    red[threadIdx.x] = ss;
    __syncthreads();
    for (int s = 128; s > 0; s >>= 1) {
        if ((int)threadIdx.x < s) red[threadIdx.x] += red[threadIdx.x + s];
        __syncthreads();
    }
    float srs = rsqrtf(red[0] / (float)(BATCH * C));
    for (int t = threadIdx.x; t < BATCH * C; t += 256) sN[t] = st[t] * srs;
    __syncthreads();

    int o = threadIdx.x;
    const float* wrow = convw + ((size_t)l * C + o) * C;
    float wss = 0.f;
    for (int i = 0; i < C; ++i) wss += wrow[i] * wrow[i];
    float wn = rsqrtf(wss / (float)C);
    for (int b = 0; b < BATCH; ++b) {
        const float* s = sN + b * C;
        float rss = 0.f;
        for (int i = 0; i < C; ++i) {
            float t = wrow[i] * wn * s[i];
            rss += t * t;
        }
        float rn = rsqrtf(rss + 1e-8f);
        _Float16* dst = wbH + ((((size_t)l * BATCH + b) * C + o) * C);
        for (int i = 0; i < C; ++i)
            dst[i] = (_Float16)(wrow[i] * wn * s[i] * rn);
    }
}

// ---------------------------------------------------------------------------
// Main per-layer GEMM + bias + LeakyReLU, f16 in / f16 out.
// Wave computes 64x64 tile; block = 8 waves = 256x128 tile.
// B slice (128 x 256 f16) staged into LDS via TDM; A streamed from global.
// A-frag (16-bit 16x32): lane m=L&15, kb=(L>>4)*16,
//   VGPR0-7 K=kb..kb+15 contiguous           -> two b128 per frag.
// B-frag (SWMMAC B table halved): lane n=L&15, kb=(L>>4)*16,
//   VGPR0-7 K=kb..kb+15 contiguous           -> two ds_load_b128 per frag.
// D: lane n=L&15, VGPR r -> m=(L>>4)*8+r.
// ---------------------------------------------------------------------------
__global__ __launch_bounds__(256) void gemm_mod_leaky(
        const _Float16* __restrict__ A,    // [B][HW][C]
        const _Float16* __restrict__ Wb,   // [B][C][C]
        const float* __restrict__ bias,    // [C]
        _Float16* __restrict__ Out) {      // [B][HW][C]
    __shared__ _Float16 Bs[BROWS * BPITCH];
    const int lane = threadIdx.x & 31;
    const int wave = threadIdx.x >> 5;
    const int wm = wave & 3, wn = wave >> 2;
    const int b = blockIdx.z;
    const int nBlock = blockIdx.x * 128;
    const int m0 = blockIdx.y * 256 + wm * 64;
    const _Float16* Ab = A + (size_t)b * HW * C;

    stage_B(Wb + (size_t)b * C * C, nBlock, Bs);

    const int ml = lane & 15;
    const int kh = (lane >> 4) * 8;
    const int kb = (lane >> 4) * 16;
    const int nloc = wn * 64;  // within LDS slice

    float8 acc[4][4] = {};

    for (int k0 = 0; k0 < C; k0 += 32) {
        half16 a[4], bf[4];
#pragma unroll
        for (int i = 0; i < 4; ++i) {
            const _Float16* ap = Ab + (size_t)(m0 + i * 16 + ml) * C + k0;
            half8 lo = *(const half8*)(ap + kb);
            half8 hi = *(const half8*)(ap + kb + 8);
            a[i] = __builtin_shufflevector(lo, hi, 0,1,2,3,4,5,6,7,8,9,10,11,12,13,14,15);
        }
#pragma unroll
        for (int j = 0; j < 4; ++j) {
            const _Float16* bp = &Bs[(nloc + j * 16 + ml) * BPITCH + k0 + kb];
            half8 lo = *(const half8*)bp;
            half8 hi = *(const half8*)(bp + 8);
            bf[j] = __builtin_shufflevector(lo, hi, 0,1,2,3,4,5,6,7,8,9,10,11,12,13,14,15);
        }
#pragma unroll
        for (int i = 0; i < 4; ++i)
#pragma unroll
            for (int j = 0; j < 4; ++j)
                acc[i][j] = __builtin_amdgcn_wmma_f32_16x16x32_f16(
                    false, a[i], false, bf[j], (short)0, acc[i][j], false, false);
    }

    const int nl = lane & 15;
    const int mh = (lane >> 4) * 8;
#pragma unroll
    for (int i = 0; i < 4; ++i)
#pragma unroll
        for (int j = 0; j < 4; ++j) {
            int ncol = nBlock + nloc + j * 16 + nl;
            float bv = bias[ncol];
            _Float16* op = Out + (size_t)b * HW * C +
                           (size_t)(m0 + i * 16 + mh) * C + ncol;
#pragma unroll
            for (int r = 0; r < 8; ++r) {
                float v = acc[i][j][r] + bv;
                v = (v >= 0.f) ? v : NEG_SLOPE * v;
                op[(size_t)r * C] = (_Float16)v;
            }
        }
}

// ---------------------------------------------------------------------------
// Final remap GEMM: f32 output, transposed store [B][COUT][HW], + bias.
// ---------------------------------------------------------------------------
__global__ __launch_bounds__(256) void gemm_remap(
        const _Float16* __restrict__ A,    // [B][HW][C]
        const _Float16* __restrict__ Wb,   // [C][C] (o-major), shared by batches
        const float* __restrict__ bias,    // [C]
        float* __restrict__ Out) {         // [B][C][HW]
    __shared__ _Float16 Bs[BROWS * BPITCH];
    const int lane = threadIdx.x & 31;
    const int wave = threadIdx.x >> 5;
    const int wm = wave & 3, wn = wave >> 2;
    const int b = blockIdx.z;
    const int nBlock = blockIdx.x * 128;
    const int m0 = blockIdx.y * 256 + wm * 64;
    const _Float16* Ab = A + (size_t)b * HW * C;

    stage_B(Wb, nBlock, Bs);

    const int ml = lane & 15;
    const int kh = (lane >> 4) * 8;
    const int kb = (lane >> 4) * 16;
    const int nloc = wn * 64;

    float8 acc[4][4] = {};

    for (int k0 = 0; k0 < C; k0 += 32) {
        half16 a[4], bf[4];
#pragma unroll
        for (int i = 0; i < 4; ++i) {
            const _Float16* ap = Ab + (size_t)(m0 + i * 16 + ml) * C + k0;
            half8 lo = *(const half8*)(ap + kb);
            half8 hi = *(const half8*)(ap + kb + 8);
            a[i] = __builtin_shufflevector(lo, hi, 0,1,2,3,4,5,6,7,8,9,10,11,12,13,14,15);
        }
#pragma unroll
        for (int j = 0; j < 4; ++j) {
            const _Float16* bp = &Bs[(nloc + j * 16 + ml) * BPITCH + k0 + kb];
            half8 lo = *(const half8*)bp;
            half8 hi = *(const half8*)(bp + 8);
            bf[j] = __builtin_shufflevector(lo, hi, 0,1,2,3,4,5,6,7,8,9,10,11,12,13,14,15);
        }
#pragma unroll
        for (int i = 0; i < 4; ++i)
#pragma unroll
            for (int j = 0; j < 4; ++j)
                acc[i][j] = __builtin_amdgcn_wmma_f32_16x16x32_f16(
                    false, a[i], false, bf[j], (short)0, acc[i][j], false, false);
    }

    const int nl = lane & 15;
    const int mh = (lane >> 4) * 8;
#pragma unroll
    for (int i = 0; i < 4; ++i)
#pragma unroll
        for (int j = 0; j < 4; ++j) {
            int ncol = nBlock + nloc + j * 16 + nl;
            float bv = bias[ncol];
            float* op = Out + (size_t)b * C * HW + (size_t)ncol * HW +
                        (m0 + i * 16 + mh);
#pragma unroll
            for (int r = 0; r < 8; ++r)
                op[r] = acc[i][j][r] + bv;
        }
}

// ---------------------------------------------------------------------------
// sigma[b][m] = | dot(base[b][m][:], sigma_w) + sigma_b |
// ---------------------------------------------------------------------------
__global__ void sigma_kernel(const _Float16* __restrict__ base, // [B][HW][C]
                             const float* __restrict__ sw,      // [C]
                             const float* __restrict__ sb,      // [1]
                             float* __restrict__ out) {         // [B*HW]
    int idx = blockIdx.x * blockDim.x + threadIdx.x;
    if (idx >= BATCH * HW) return;
    const _Float16* p = base + (size_t)idx * C;
    float acc = 0.f;
    for (int c = 0; c < C; c += 8) {
        half8 v = *(const half8*)(p + c);
#pragma unroll
        for (int j = 0; j < 8; ++j) acc += (float)v[j] * sw[c + j];
    }
    acc += sb[0];
    out[idx] = fabsf(acc);
}

// ---------------------------------------------------------------------------
// Launch
// ---------------------------------------------------------------------------
extern "C" void kernel_launch(void* const* d_in, const int* in_sizes, int n_in,
                              void* d_out, int out_size, void* d_ws, size_t ws_size,
                              hipStream_t stream) {
    const float* x        = (const float*)d_in[0];  // [B,H,W,CIN]
    const float* wsv      = (const float*)d_in[1];  // [D,B,WDIM]
    const float* affine_w = (const float*)d_in[2];  // [D,CIN,WDIM]
    const float* affine_b = (const float*)d_in[3];  // [D,CIN]
    const float* conv_w   = (const float*)d_in[4];  // [D,COUT,CIN]
    const float* conv_b   = (const float*)d_in[5];  // [D,COUT]
    const float* sigma_w  = (const float*)d_in[6];  // [1,COUT]
    const float* sigma_b  = (const float*)d_in[7];  // [1]
    const float* remap_w  = (const float*)d_in[8];  // [COUT,COUT]
    const float* remap_b  = (const float*)d_in[9];  // [COUT]
    float* out = (float*)d_out;

    // workspace layout
    char* ws = (char*)d_ws;
    const size_t actElems = (size_t)BATCH * HW * C;            // 16,777,216
    _Float16* actX  = (_Float16*)ws;                           // 32 MB
    size_t off = actElems * sizeof(_Float16);
    float* styles   = (float*)(ws + off);                      // 32 KB
    off += (size_t)D_LAYERS * BATCH * C * sizeof(float);
    _Float16* wbH   = (_Float16*)(ws + off);                   // 4 MB
    off += (size_t)D_LAYERS * BATCH * C * C * sizeof(_Float16);
    _Float16* remapH = (_Float16*)(ws + off);                  // 128 KB

    // second ping buffer borrows the front of d_out (fully overwritten later)
    _Float16* actY = (_Float16*)d_out;

    // P0: x -> f16
    {
        int n = (int)actElems;
        cvt_f32_to_f16<<<(n + 255) / 256, 256, 0, stream>>>(x, actX, n);
    }
    // P1: styles
    {
        int n = D_LAYERS * BATCH * C;
        styles_kernel<<<(n + 255) / 256, 256, 0, stream>>>(wsv, affine_w, affine_b, styles);
    }
    // P2: modulated weights
    modweights_kernel<<<D_LAYERS, 256, 0, stream>>>(styles, conv_w, wbH);
    // P3: remap weights -> f16
    {
        int n = C * C;
        cvt_f32_to_f16<<<(n + 255) / 256, 256, 0, stream>>>(remap_w, remapH, n);
    }

    // 8 modulated-conv layers (WMMA GEMMs), ping-pong X <-> Y
    dim3 grid(C / 128, HW / 256, BATCH);  // (2, 64, 4)
    for (int l = 0; l < D_LAYERS; ++l) {
        const _Float16* in   = (l & 1) ? actY : actX;
        _Float16*       outp = (l & 1) ? actX : actY;
        gemm_mod_leaky<<<grid, 256, 0, stream>>>(
            in, wbH + (size_t)l * BATCH * C * C, conv_b + (size_t)l * C, outp);
    }
    // after 8 layers result is in actX

    // final remap -> feat [B,COUT,H,W] in d_out (f32)
    gemm_remap<<<grid, 256, 0, stream>>>(actX, remapH, remap_b, out);

    // sigma -> appended after feat
    {
        float* sigma_out = out + (size_t)BATCH * C * HW;
        int n = BATCH * HW;
        sigma_kernel<<<(n + 255) / 256, 256, 0, stream>>>(actX, sigma_w, sigma_b, sigma_out);
    }
}
